// SparseAttention_72688026517861
// MI455X (gfx1250) — matrix operand order repaired and checked by reading, not compiled, not run on hardware
//
#include <hip/hip_runtime.h>
#include <hip/hip_bf16.h>

// ---------------------------------------------------------------------------
// Masked MHA (B=2, N=2304, C=256, h=8, d=32) via CDNA5 bf16 WMMA.
// All GEMM stages use V_WMMA_F32_16X16X32_BF16 (K=32/instr, f32 accum).
// Flash-attention style online softmax keeps scores in registers; P tile is
// relaid out C-layout -> A-layout through LDS (ds_store/ds_load + s_wait_dscnt).
// ---------------------------------------------------------------------------

typedef __attribute__((ext_vector_type(16))) __bf16 v16bf;
typedef __attribute__((ext_vector_type(8)))  __bf16 v8bf;
typedef __attribute__((ext_vector_type(8)))  float  v8f;

union V16 { v16bf v; v8bf h[2]; };

#define NB   2
#define NH   8
#define NN   2304      // 48*48
#define NC   256
#define ND   32
#define NW   48
#define NT   144       // N/16 tiles

__device__ __forceinline__ v8f wmma_bf16(v16bf a, v16bf b, v8f c) {
  // (neg_a, A, neg_b, B, c_mod, C, reuse_a, reuse_b)
  return __builtin_amdgcn_wmma_f32_16x16x32_bf16(false, a, false, b, (short)0, c,
                                                 false, false);
}

// ---- kernel 0: x[B,C,N] f32 -> Xt[B,N,C] bf16 ------------------------------
__global__ __launch_bounds__(256) void k_transpose(const float* __restrict__ x,
                                                   __bf16* __restrict__ Xt) {
  int idx = blockIdx.x * 256 + threadIdx.x;           // over B*C*N
  if (idx >= NB * NC * NN) return;
  int n = idx % NN;
  int t = idx / NN;
  int c = t % NC;
  int b = t / NC;
  Xt[((size_t)b * NN + n) * NC + c] = (__bf16)x[idx];
}

// ---- kernel 1: qkv = Xt @ w_in^T + b_in; scatter to Q/K/Vt -----------------
__global__ __launch_bounds__(256) void k_qkv(const __bf16* __restrict__ Xt,
                                             const float* __restrict__ w_in,
                                             const float* __restrict__ b_in,
                                             __bf16* __restrict__ Qb,
                                             __bf16* __restrict__ Kb,
                                             __bf16* __restrict__ Vt) {
  int wave = (blockIdx.x * 256 + threadIdx.x) >> 5;   // 0 .. B*NT*48-1
  if (wave >= NB * NT * 48) return;
  int lane = threadIdx.x & 31, lo = lane & 15, hi = lane >> 4;
  int mt = wave % 48;
  int t  = wave / 48;
  int nt = t % NT;
  int b  = t / NT;
  int n0 = nt * 16, m0 = mt * 16;
  int m  = m0 + lo;                                   // this lane's output col

  v8f acc = {};
  const __bf16* arow = Xt + ((size_t)b * NN + n0 + lo) * NC;
  const float*  wrow = w_in + (size_t)m * NC;
  for (int ks = 0; ks < 8; ++ks) {
    int cb = ks * 32;
    V16 a;
    a.h[0] = *(const v8bf*)(arow + cb + hi * 16);      // K = hi*16 .. +7
    a.h[1] = *(const v8bf*)(arow + cb + hi * 16 + 8);  // K = hi*16+8 .. +15
    V16 bb;
    const float* wp = wrow + cb + hi * 16;             // K = hi*16 + j
#pragma unroll
    for (int j = 0; j < 16; ++j) bb.v[j] = (__bf16)wp[j];
    acc = wmma_bf16(a.v, bb.v, acc);
  }

  float bias  = b_in[m];
  int   sec   = m >> 8;                                // 0:q 1:k 2:v
  int   cl    = m & 255;
  int   h     = cl >> 5, d = cl & 31;
  float scale = (sec == 0) ? 0.17677669529663687f : 1.0f;  // 1/sqrt(32) on Q
#pragma unroll
  for (int r = 0; r < 8; ++r) {
    int n = n0 + r + 8 * hi;
    __bf16 bv = (__bf16)((acc[r] + bias) * scale);
    if (sec == 0)      Qb[((size_t)(b * NH + h) * NN + n) * ND + d] = bv;
    else if (sec == 1) Kb[((size_t)(b * NH + h) * NN + n) * ND + d] = bv;
    else               Vt[((size_t)(b * NH + h) * ND + d) * NN + n] = bv;
  }
}

// ---- kernel 2: flash attention, one 16-query tile per wave -----------------
__global__ __launch_bounds__(256) void k_attn(const __bf16* __restrict__ Qb,
                                              const __bf16* __restrict__ Kb,
                                              const __bf16* __restrict__ Vt,
                                              __bf16* __restrict__ Obf) {
  __shared__ __bf16 lds[8][16 * 32];                   // per-wave P tile
  int wslot = threadIdx.x >> 5;
  int wave  = blockIdx.x * 8 + wslot;                  // 0 .. B*NH*NT-1
  int lane  = threadIdx.x & 31, lo = lane & 15, hi = lane >> 4;
  int qt = wave % NT;
  int t  = wave / NT;
  int h  = t % NH;
  int b  = t / NH;
  int bh = b * NH + h;
  int q0 = qt * 16;

  // Q fragment (A-layout), loaded once
  V16 aq;
  const __bf16* qp = Qb + ((size_t)bh * NN + q0 + lo) * ND;
  aq.h[0] = *(const v8bf*)(qp + hi * 16);
  aq.h[1] = *(const v8bf*)(qp + hi * 16 + 8);

  float m_i[8], l_i[8];
  v8f olo = {}, ohi = {};
#pragma unroll
  for (int r = 0; r < 8; ++r) { m_i[r] = -1e30f; l_i[r] = 0.0f; }

  v8f zero = {};
  for (int kt = 0; kt < NT; kt += 2) {
    int kb0 = kt * 16;                                 // 32 keys per iter
    const __bf16* kp = Kb + ((size_t)bh * NN + kb0 + lo) * ND + hi * 16;
    v16bf bk0 = *(const v16bf*)kp;                     // keys kb0..+15
    v16bf bk1 = *(const v16bf*)(kp + 16 * ND);         // keys kb0+16..+31
    if (kt + 2 < NT)                                   // global_prefetch next K
      __builtin_prefetch(Kb + ((size_t)bh * NN + (kt + 2) * 16 + lo) * ND, 0, 0);

    v8f s0 = wmma_bf16(aq.v, bk0, zero);
    v8f s1 = wmma_bf16(aq.v, bk1, zero);

    int k0 = kb0 + lo;       int k0i = k0 / NW, k0j = k0 - k0i * NW;
    int k1 = k0 + 16;        int k1i = k1 / NW, k1j = k1 - k1i * NW;

#pragma unroll
    for (int r = 0; r < 8; ++r) {
      int qn = q0 + r + 8 * hi;
      int qi = qn / NW, qj = qn - qi * NW;
      int d0i = qi - k0i; d0i = d0i < 0 ? -d0i : d0i;
      int d0j = qj - k0j; d0j = d0j < 0 ? -d0j : d0j;
      int d1i = qi - k1i; d1i = d1i < 0 ? -d1i : d1i;
      int d1j = qj - k1j; d1j = d1j < 0 ? -d1j : d1j;
      // reference quirk: 3x3 neighborhood (incl. self) is BLOCKED
      float v0 = (d0i <= 1 && d0j <= 1) ? -1e30f : s0[r];
      float v1 = (d1i <= 1 && d1j <= 1) ? -1e30f : s1[r];

      float mx = fmaxf(v0, v1);
      mx = fmaxf(mx, __shfl_xor(mx, 1, 32));
      mx = fmaxf(mx, __shfl_xor(mx, 2, 32));
      mx = fmaxf(mx, __shfl_xor(mx, 4, 32));
      mx = fmaxf(mx, __shfl_xor(mx, 8, 32));
      float mnew = fmaxf(m_i[r], mx);
      float sc   = __expf(m_i[r] - mnew);
      float p0   = __expf(v0 - mnew);
      float p1   = __expf(v1 - mnew);
      float rs   = p0 + p1;
      rs += __shfl_xor(rs, 1, 32);
      rs += __shfl_xor(rs, 2, 32);
      rs += __shfl_xor(rs, 4, 32);
      rs += __shfl_xor(rs, 8, 32);
      l_i[r] = l_i[r] * sc + rs;
      m_i[r] = mnew;
      olo[r] *= sc;
      ohi[r] *= sc;
      int mr = r + 8 * hi;                              // C-layout row
      lds[wslot][mr * 32 + lo]      = (__bf16)p0;
      lds[wslot][mr * 32 + 16 + lo] = (__bf16)p1;
    }
    asm volatile("s_wait_dscnt 0" ::: "memory");        // P visible to all lanes

    V16 ap;                                             // P as A-fragment
    ap.h[0] = *(const v8bf*)&lds[wslot][lo * 32 + hi * 16];
    ap.h[1] = *(const v8bf*)&lds[wslot][lo * 32 + hi * 16 + 8];

    const __bf16* vp0 = Vt + ((size_t)bh * ND + lo) * NN + kb0 + hi * 16;
    const __bf16* vp1 = Vt + ((size_t)bh * ND + 16 + lo) * NN + kb0 + hi * 16;
    v16bf bv0 = *(const v16bf*)vp0;                     // d = 0..15 columns
    v16bf bv1 = *(const v16bf*)vp1;                     // d = 16..31 columns
    olo = wmma_bf16(ap.v, bv0, olo);
    ohi = wmma_bf16(ap.v, bv1, ohi);
  }

#pragma unroll
  for (int r = 0; r < 8; ++r) {
    float inv = 1.0f / l_i[r];
    int n = q0 + r + 8 * hi;
    size_t base = ((size_t)b * NN + n) * NC + h * ND;
    Obf[base + lo]      = (__bf16)(olo[r] * inv);
    Obf[base + 16 + lo] = (__bf16)(ohi[r] * inv);
  }
}

// ---- kernel 3: out = Obf @ w_out^T + b_out, write [B,C,H,W] f32 ------------
__global__ __launch_bounds__(256) void k_outproj(const __bf16* __restrict__ Obf,
                                                 const float* __restrict__ w_out,
                                                 const float* __restrict__ b_out,
                                                 float* __restrict__ out) {
  int wave = (blockIdx.x * 256 + threadIdx.x) >> 5;     // 0 .. B*NT*16-1
  if (wave >= NB * NT * 16) return;
  int lane = threadIdx.x & 31, lo = lane & 15, hi = lane >> 4;
  int ct = wave % 16;
  int t  = wave / 16;
  int nt = t % NT;
  int b  = t / NT;
  int c0 = ct * 16, n0 = nt * 16;

  v8f acc = {};
  const float*  wrow = w_out + (size_t)(c0 + lo) * NC;  // A rows = c_out
  const __bf16* orow = Obf + ((size_t)b * NN + n0 + lo) * NC;  // B cols = n
  for (int ks = 0; ks < 8; ++ks) {
    int cb = ks * 32;
    V16 a;
#pragma unroll
    for (int j = 0; j < 8; ++j) {
      a.v[j]     = (__bf16)wrow[cb + hi * 16 + j];
      a.v[8 + j] = (__bf16)wrow[cb + hi * 16 + 8 + j];
    }
    v16bf bo = *(const v16bf*)(orow + cb + hi * 16);
    acc = wmma_bf16(a.v, bo, acc);
  }
#pragma unroll
  for (int r = 0; r < 8; ++r) {
    int c = c0 + r + 8 * hi;
    int n = n0 + lo;
    out[((size_t)b * NC + c) * NN + n] = acc[r] + b_out[c];
  }
}

// ---------------------------------------------------------------------------
extern "C" void kernel_launch(void* const* d_in, const int* in_sizes, int n_in,
                              void* d_out, int out_size, void* d_ws, size_t ws_size,
                              hipStream_t stream) {
  const float* x     = (const float*)d_in[0];
  const float* w_in  = (const float*)d_in[1];
  const float* b_in  = (const float*)d_in[2];
  const float* w_out = (const float*)d_in[3];
  const float* b_out = (const float*)d_in[4];
  float* out = (float*)d_out;

  // workspace layout (bf16, each region 2,359,296 bytes; total ~11.25 MB)
  const size_t REG = (size_t)NB * NN * NC * sizeof(__hip_bfloat16); // 2359296
  char* ws = (char*)d_ws;
  __bf16* Xt  = (__bf16*)(ws + 0 * REG);   // [B][N][C]
  __bf16* Qb  = (__bf16*)(ws + 1 * REG);   // [B][H][N][D]  (pre-scaled)
  __bf16* Kb  = (__bf16*)(ws + 2 * REG);   // [B][H][N][D]
  __bf16* Vt  = (__bf16*)(ws + 3 * REG);   // [B][H][D][N]  (transposed)
  __bf16* Obf = (__bf16*)(ws + 4 * REG);   // [B][N][C]

  k_transpose<<<(NB * NC * NN + 255) / 256, 256, 0, stream>>>(x, Xt);
  k_qkv<<<(NB * NT * 48) / 8, 256, 0, stream>>>(Xt, w_in, b_in, Qb, Kb, Vt);
  k_attn<<<(NB * NH * NT) / 8, 256, 0, stream>>>(Qb, Kb, Vt, Obf);
  k_outproj<<<(NB * NT * 16) / 8, 256, 0, stream>>>(Obf, w_out, b_out, out);
}
